// SoftmaxAttention_43593918054530
// MI455X (gfx1250) — compile-verified, matrix operand order checked by reading, not run on hardware
//
#include <hip/hip_runtime.h>
#include <hip/hip_bf16.h>

// MI455X / gfx1250, wave32. bf16 WMMA (16x16x32, f32 accum) everywhere.
// Fragments come from K-contiguous LDS/global tiles via 2x b128 loads.
// Attention K tiles move via the Tensor Data Mover, double-buffered so the
// DMA overlaps the WMMAs of the previous tile.

typedef __attribute__((ext_vector_type(16))) __bf16 v16bf;
typedef __attribute__((ext_vector_type(8)))  __bf16 v8bf;
typedef __attribute__((ext_vector_type(4)))  __bf16 v4bf;
typedef __attribute__((ext_vector_type(8)))  float  v8f;
typedef __attribute__((ext_vector_type(4)))  float  v4f;

#if defined(__has_builtin)
#if __has_builtin(__builtin_amdgcn_tensor_load_to_lds) && \
    __has_builtin(__builtin_amdgcn_s_wait_tensorcnt)
#define HAVE_TDM 1
#endif
#endif

#define B_   2
#define N_   1024
#define D_   2048
#define H_   8
#define I_   4
#define KQD  128
#define VD   128
#define BN   (B_*N_)          // 2048 token rows
#define HIK  (H_*I_*KQD)      // 4096
#define HK   (H_*KQD)         // 1024
#define HV   (H_*VD)          // 1024
#define HIV  (H_*I_*VD)       // 4096

// 16-bit A/B fragment: lane (m,hi) holds K runs [hi*8, hi*8+8) and
// [16+hi*8, 16+hi*8+8) of its row/column -> two 16-byte loads.
__device__ __forceinline__ v16bf load_frag(const __bf16* base, int hi) {
  v8bf lo  = *(const v8bf*)(base + hi * 8);
  v8bf hi8 = *(const v8bf*)(base + 16 + hi * 8);
  return __builtin_shufflevector(lo, hi8, 0, 1, 2, 3, 4, 5, 6, 7,
                                 8, 9, 10, 11, 12, 13, 14, 15);
}

// ---------------------------------------------------------------------------
// C[M,N] = A[M,K](f32) x B[K,N](f32), bf16 WMMA, f32 out.
// 4 waves / block, 64x64 tile, K stepped by 32, software-pipelined with
// running per-thread pointers (no 64-bit muls in the hot loop).
// ---------------------------------------------------------------------------
#define TPAD 40

__global__ __launch_bounds__(128)
void gemm_f32_bf16wmma(const float* __restrict__ A, const float* __restrict__ Bm,
                       float* __restrict__ C, int Ncols, int Kd) {
  __shared__ __bf16 As[64 * TPAD];   // [row][k]
  __shared__ __bf16 Bt[64 * TPAD];   // [col][k]
  const int tid  = threadIdx.x;
  const int lane = tid & 31;
  const int wave = tid >> 5;
  const int m    = lane & 15;
  const int hi   = lane >> 4;
  const int row0 = blockIdx.y * 64;
  const int col0 = blockIdx.x * 64;

  const float* aPtr[4];
  const float* bPtr[4];
#pragma unroll
  for (int it = 0; it < 4; ++it) {
    int idx = it * 128 + tid;             // 0..511
    aPtr[it] = A + (size_t)(row0 + (idx >> 3)) * Kd + (idx & 7) * 4;
    bPtr[it] = Bm + (size_t)(idx >> 4) * Ncols + col0 + (idx & 15) * 4;
  }
  const size_t bstep = (size_t)32 * Ncols;

  v4f a_reg[4], b_reg[4];
  auto load_regs = [&]() {
#pragma unroll
    for (int it = 0; it < 4; ++it) a_reg[it] = *(const v4f*)aPtr[it];
#pragma unroll
    for (int it = 0; it < 4; ++it) b_reg[it] = *(const v4f*)bPtr[it];
  };
  auto advance = [&]() {
#pragma unroll
    for (int it = 0; it < 4; ++it) { aPtr[it] += 32; bPtr[it] += bstep; }
  };

  v8f acc[4] = {};
  load_regs();
  if (32 < Kd) advance();
  for (int k0 = 0; k0 < Kd; k0 += 32) {
    // Store staged registers (f32 -> bf16) into LDS
#pragma unroll
    for (int it = 0; it < 4; ++it) {
      int idx = it * 128 + tid;
      int r = idx >> 3, c4 = idx & 7;
      v4bf p; p[0] = (__bf16)a_reg[it][0]; p[1] = (__bf16)a_reg[it][1];
      p[2] = (__bf16)a_reg[it][2]; p[3] = (__bf16)a_reg[it][3];
      *(v4bf*)(As + r * TPAD + c4 * 4) = p;
    }
#pragma unroll
    for (int it = 0; it < 4; ++it) {
      int idx = it * 128 + tid;
      int k = idx >> 4, c4 = idx & 15;
#pragma unroll
      for (int u = 0; u < 4; ++u)
        Bt[(c4 * 4 + u) * TPAD + k] = (__bf16)b_reg[it][u];
    }
    __syncthreads();

    // Prefetch next K-step from global while WMMAs run
    load_regs();
    if (k0 + 64 < Kd) advance();

    v16bf af  = load_frag(&As[(wave * 16 + m) * TPAD], hi);
    v16bf bf0 = load_frag(&Bt[(0 * 16 + m) * TPAD], hi);
    v16bf bf1 = load_frag(&Bt[(1 * 16 + m) * TPAD], hi);
    v16bf bf2 = load_frag(&Bt[(2 * 16 + m) * TPAD], hi);
    v16bf bf3 = load_frag(&Bt[(3 * 16 + m) * TPAD], hi);
    acc[0] = __builtin_amdgcn_wmma_f32_16x16x32_bf16(false, af, false, bf0,
                                                     (short)0, acc[0], false, false);
    acc[1] = __builtin_amdgcn_wmma_f32_16x16x32_bf16(false, af, false, bf1,
                                                     (short)0, acc[1], false, false);
    acc[2] = __builtin_amdgcn_wmma_f32_16x16x32_bf16(false, af, false, bf2,
                                                     (short)0, acc[2], false, false);
    acc[3] = __builtin_amdgcn_wmma_f32_16x16x32_bf16(false, af, false, bf3,
                                                     (short)0, acc[3], false, false);
    __syncthreads();
  }

#pragma unroll
  for (int j = 0; j < 4; ++j)
#pragma unroll
    for (int jj = 0; jj < 8; ++jj)
      C[(size_t)(row0 + wave * 16 + jj + hi * 8) * Ncols + col0 + j * 16 + m] =
          acc[j][jj];
}

// ---------------------------------------------------------------------------
// RoPE over f32 rows of width `cols` (multiple of 128), write bf16.
// ---------------------------------------------------------------------------
__global__ void rope_bf16(const float* __restrict__ in, __bf16* __restrict__ out,
                          int cols, float theta) {
  int idx = blockIdx.x * blockDim.x + threadIdx.x;
  int pairsPerRow = cols >> 1;
  int total = BN * pairsPerRow;
  if (idx >= total) return;
  int row = idx / pairsPerRow;
  int rem = idx - row * pairsPerRow;
  int blk = rem >> 6;
  int j   = rem & 63;
  int c1 = blk * 128 + j;
  int c2 = c1 + 64;
  int n  = row & (N_ - 1);
  float freq = __powf(theta, -(float)(2 * j) * (1.0f / 128.0f));
  float ang = (float)n * freq;
  float cs = __cosf(ang), sn = __sinf(ang);
  float x1 = in[(size_t)row * cols + c1];
  float x2 = in[(size_t)row * cols + c2];
  out[(size_t)row * cols + c1] = (__bf16)(x1 * cs - x2 * sn);
  out[(size_t)row * cols + c2] = (__bf16)(x1 * sn + x2 * cs);
}

__global__ void cvt_bf16(const float* __restrict__ in, __bf16* __restrict__ out, int n) {
  int i = blockIdx.x * blockDim.x + threadIdx.x;
  if (i < n) out[i] = (__bf16)in[i];
}

// ---------------------------------------------------------------------------
// Tensor Data Mover: load a 32x128 bf16 tile (rows strided HK elements) into
// LDS, hardware-padding each 64-DWORD row by 4 DWORDs -> 136-element rows.
// 6-arg builtin: (g0 u32x4, g1 i32x8, g2 i32x4, g3 i32x4, g2b i32x8, cpol).
// ---------------------------------------------------------------------------
#ifdef HAVE_TDM
typedef __attribute__((ext_vector_type(4))) unsigned int u32x4;
typedef __attribute__((ext_vector_type(8))) int i32x8;
typedef __attribute__((ext_vector_type(4))) int i32x4;

__device__ __forceinline__ void tdm_load_k_tile(const __bf16* gsrc,
                                                unsigned int lds_byte_off) {
  unsigned long long ga = (unsigned long long)(uintptr_t)gsrc;
  u32x4 g0 = {
    1u,                                              // count=1 user descriptor
    lds_byte_off,                                    // LDS destination
    (unsigned int)(ga & 0xffffffffu),                // global_addr[31:0]
    (unsigned int)((ga >> 32) & 0x1ffffffu) | (2u << 30) // addr[56:32] | type=2
  };
  i32x8 g1 = {
    (int)((1u << 16)        // data_size = 2 bytes
        | (1u << 20)        // pad_enable
        | (5u << 22)        // pad_interval: 64 DWORDs (one 128-elem row)
        | (3u << 25)),      // pad_amount: 4 DWORDs (8 bf16)
    (int)((128u & 0xffffu) << 16),                   // tensor_dim0 lo16
    (int)(((1024u & 0xffffu) << 16)),                // dim0 hi16=0 | tensor_dim1 lo16
    (int)(((1024u >> 16) & 0xffffu) | (128u << 16)), // dim1 hi16 | tile_dim0=128
    (int)32u,                                        // tile_dim1=32 | tile_dim2=0
    (int)HK,                                         // tensor_dim0_stride = 1024
    0, 0                                             // dim1_stride unused (2D)
  };
  i32x4 z4 = {};
  i32x8 z8 = {};
  __builtin_amdgcn_tensor_load_to_lds(g0, g1, z4, z4, z8, 0);
}
#endif

// ---------------------------------------------------------------------------
// Causal flash attention. One 4-wave block per (b, h, 16-row q block);
// wave w = query head i, sharing staged K/V. Double-buffered tiles:
// TDM (K) and transpose stores (V) for tile i+1 overlap tile i's WMMAs.
// ---------------------------------------------------------------------------
#define KSTRIDE 136    // 128 + 8 pad, rows 272B (16B aligned)
#define VSTRIDE 40     // 32 + 8 pad, rows 80B (16B aligned)

__global__ __launch_bounds__(128)
void attn_wmma(const __bf16* __restrict__ qb, const __bf16* __restrict__ kb,
               const __bf16* __restrict__ vb, float* __restrict__ attn) {
  __shared__ __bf16 Klds[2][32 * KSTRIDE];  // [buf][key][dim]
  __shared__ __bf16 Vt[2][128 * VSTRIDE];   // [buf][vd][key]
  __shared__ __bf16 Plds[4][16 * 32];       // per-wave P tile [qrow][key]

  const int tid  = threadIdx.x;
  const int lane = tid & 31;
  const int wave = tid >> 5;                // = query head i
  const int m    = lane & 15;
  const int hi   = lane >> 4;

  int bid = blockIdx.x;
  const int qblk = bid & 63; bid >>= 6;
  const int h    = bid & 7;  bid >>= 3;
  const int b    = bid;
  const int q0   = qblk * 16;

  const size_t qbase = (size_t)(b * N_) * HIK + (size_t)h * (I_ * KQD) + (size_t)wave * KQD;
  const size_t kbase = (size_t)(b * N_) * HK + (size_t)h * KQD;
  const size_t vbase = (size_t)(b * N_) * HV + (size_t)h * VD;

  // Q fragments: rows are d-contiguous in global -> direct b128 loads
  v16bf aq[4];
  {
    const __bf16* qrow = qb + qbase + (size_t)(q0 + m) * HIK;
#pragma unroll
    for (int c = 0; c < 4; ++c)
      aq[c] = load_frag(qrow + c * 32, hi);
  }

  auto stage_v = [&](int buf, int k0) {
#pragma unroll
    for (int it = 0; it < 4; ++it) {
      int idx = it * 128 + tid;             // 32 rows x 16 chunks of 8
      int rr = idx >> 4, cc = idx & 15;
      v8bf vv = *(const v8bf*)(vb + vbase + (size_t)(k0 + rr) * HV + cc * 8);
#pragma unroll
      for (int u = 0; u < 8; ++u)
        Vt[buf][(cc * 8 + u) * VSTRIDE + rr] = vv[u];
    }
  };
#ifndef HAVE_TDM
  auto stage_k = [&](int buf, int k0) {
#pragma unroll
    for (int it = 0; it < 4; ++it) {
      int idx = it * 128 + tid;
      int rr = idx >> 4, cc = idx & 15;
      v8bf kv = *(const v8bf*)(kb + kbase + (size_t)(k0 + rr) * HK + cc * 8);
      *(v8bf*)(&Klds[buf][rr * KSTRIDE + cc * 8]) = kv;
    }
  };
#endif

  float mrow[8], lrow[8];
  v8f   O[8];
#pragma unroll
  for (int j = 0; j < 8; ++j) { mrow[j] = -3.0e38f; lrow[j] = 0.0f; }
#pragma unroll
  for (int t = 0; t < 8; ++t) O[t] = (v8f){};

  const float scale = 0.08838834764831845f;   // 1/sqrt(128)
  const int nkb = (q0 + 47) >> 5;             // 32-key blocks with k0 <= q0+15

  // Prologue: stage tile 0 into buffer 0
  stage_v(0, 0);
#ifdef HAVE_TDM
  if (wave == 0) {
    tdm_load_k_tile(kb + kbase, (unsigned int)(uintptr_t)&Klds[0][0]);
    __builtin_amdgcn_s_wait_tensorcnt(0);
  }
#else
  stage_k(0, 0);
#endif
  __syncthreads();

  for (int kbk = 0; kbk < nkb; ++kbk) {
    const int k0  = kbk * 32;
    const int cur = kbk & 1;
    const bool hasNext = (kbk + 1 < nkb);

    // Kick off next tile's staging before computing on the current one
    if (hasNext) {
      if (k0 + 64 < N_)
        __builtin_prefetch(vb + vbase + (size_t)(k0 + 64) * HV, 0, 1);
      stage_v(cur ^ 1, k0 + 32);
#ifdef HAVE_TDM
      if (wave == 0)
        tdm_load_k_tile(kb + kbase + (size_t)(k0 + 32) * HK,
                        (unsigned int)(uintptr_t)&Klds[cur ^ 1][0]);
#else
      stage_k(cur ^ 1, k0 + 32);
#endif
    }

    // Scores: two 16x16 tiles over keys k0..k0+31 (paired for independence)
    v8f S[2] = {};
#pragma unroll
    for (int c = 0; c < 4; ++c) {
      v16bf bk0 = load_frag(&Klds[cur][(0 * 16 + m) * KSTRIDE + c * 32], hi);
      v16bf bk1 = load_frag(&Klds[cur][(1 * 16 + m) * KSTRIDE + c * 32], hi);
      S[0] = __builtin_amdgcn_wmma_f32_16x16x32_bf16(false, aq[c], false, bk0,
                                                     (short)0, S[0], false, false);
      S[1] = __builtin_amdgcn_wmma_f32_16x16x32_bf16(false, aq[c], false, bk1,
                                                     (short)0, S[1], false, false);
    }

    // Scale + causal mask + online softmax (rows live across 16-lane halves)
#pragma unroll
    for (int j = 0; j < 8; ++j) {
      int   qr = q0 + j + hi * 8;
      float s0 = S[0][j] * scale;
      float s1 = S[1][j] * scale;
      if (k0 + m      > qr) s0 = -3.0e38f;
      if (k0 + 16 + m > qr) s1 = -3.0e38f;
      float mx = fmaxf(s0, s1);
#pragma unroll
      for (int off = 1; off < 16; off <<= 1)
        mx = fmaxf(mx, __shfl_xor(mx, off, 32));
      float mn = fmaxf(mrow[j], mx);
      float p0 = __expf(s0 - mn);
      float p1 = __expf(s1 - mn);
      Plds[wave][(j + hi * 8) * 32 + m]      = (__bf16)p0;
      Plds[wave][(j + hi * 8) * 32 + 16 + m] = (__bf16)p1;
      float ps = p0 + p1;
#pragma unroll
      for (int off = 1; off < 16; off <<= 1)
        ps += __shfl_xor(ps, off, 32);
      float alpha = __expf(mrow[j] - mn);
      lrow[j] = lrow[j] * alpha + ps;
      mrow[j] = mn;
#pragma unroll
      for (int t = 0; t < 8; ++t) O[t][j] *= alpha;
    }
    __syncthreads();   // publish P transpose (and keep waves loosely in step)

    // PV: P rows key-contiguous (A side), Vt rows key-contiguous (B side)
    v16bf ap = load_frag(&Plds[wave][m * 32], hi);
#pragma unroll
    for (int vt = 0; vt < 8; vt += 2) {
      v16bf bv0 = load_frag(&Vt[cur][(vt * 16 + m) * VSTRIDE], hi);
      v16bf bv1 = load_frag(&Vt[cur][((vt + 1) * 16 + m) * VSTRIDE], hi);
      O[vt]     = __builtin_amdgcn_wmma_f32_16x16x32_bf16(false, ap, false, bv0,
                                                          (short)0, O[vt], false, false);
      O[vt + 1] = __builtin_amdgcn_wmma_f32_16x16x32_bf16(false, ap, false, bv1,
                                                          (short)0, O[vt + 1], false, false);
    }

    // Make sure next tile's K DMA landed, then publish both next buffers
#ifdef HAVE_TDM
    if (hasNext && wave == 0) __builtin_amdgcn_s_wait_tensorcnt(0);
#endif
    __syncthreads();
  }

  // Normalize and write [b, n, h, i, v] (f32) for the output GEMM
#pragma unroll
  for (int j = 0; j < 8; ++j) {
    float inv = 1.0f / lrow[j];
#pragma unroll
    for (int vt = 0; vt < 8; ++vt)
      attn[(size_t)(b * N_ + q0 + j + hi * 8) * HIV +
           (size_t)h * (I_ * VD) + (size_t)wave * VD + vt * 16 + m] = O[vt][j] * inv;
  }
}

// ---------------------------------------------------------------------------
extern "C" void kernel_launch(void* const* d_in, const int* in_sizes, int n_in,
                              void* d_out, int out_size, void* d_ws, size_t ws_size,
                              hipStream_t stream) {
  (void)in_sizes; (void)n_in; (void)out_size; (void)ws_size;
  const float* x  = (const float*)d_in[0];
  const float* Wq = (const float*)d_in[1];
  const float* Wk = (const float*)d_in[2];
  const float* Wv = (const float*)d_in[3];
  const float* Wo = (const float*)d_in[4];
  float* out = (float*)d_out;

  char* ws = (char*)d_ws;
  size_t off = 0;
  float*  qf    = (float*)(ws + off);  off += (size_t)BN * HIK * sizeof(float);
  float*  kf    = (float*)(ws + off);  off += (size_t)BN * HK  * sizeof(float);
  float*  vf    = (float*)(ws + off);  off += (size_t)BN * HV  * sizeof(float);
  __bf16* qbf   = (__bf16*)(ws + off); off += (size_t)BN * HIK * sizeof(__bf16);
  __bf16* kbf   = (__bf16*)(ws + off); off += (size_t)BN * HK  * sizeof(__bf16);
  __bf16* vbf   = (__bf16*)(ws + off); off += (size_t)BN * HV  * sizeof(__bf16);
  float*  attnf = (float*)(ws + off);  off += (size_t)BN * HIV * sizeof(float);

  dim3 blk(128);
  // Projections (64x64 tiles)
  gemm_f32_bf16wmma<<<dim3(HIK / 64, BN / 64), blk, 0, stream>>>(x, Wq, qf, HIK, D_);
  gemm_f32_bf16wmma<<<dim3(HK  / 64, BN / 64), blk, 0, stream>>>(x, Wk, kf, HK,  D_);
  gemm_f32_bf16wmma<<<dim3(HV  / 64, BN / 64), blk, 0, stream>>>(x, Wv, vf, HV,  D_);
  // RoPE + bf16 conversion
  {
    int total = BN * (HIK / 2);
    rope_bf16<<<(total + 255) / 256, 256, 0, stream>>>(qf, qbf, HIK, 10000.0f);
  }
  {
    int total = BN * (HK / 2);
    rope_bf16<<<(total + 255) / 256, 256, 0, stream>>>(kf, kbf, HK, 10000.0f);
  }
  {
    int total = BN * HV;
    cvt_bf16<<<(total + 255) / 256, 256, 0, stream>>>(vf, vbf, total);
  }
  // Flash attention: (b, h, qblock) blocks, 4 waves = 4 query heads
  attn_wmma<<<dim3(B_ * H_ * (N_ / 16)), blk, 0, stream>>>(qbf, kbf, vbf, attnf);
  // Output projection
  gemm_f32_bf16wmma<<<dim3(D_ / 64, BN / 64), blk, 0, stream>>>(attnf, Wo, out, D_, HIV);
}
